// SelfAttention_84327387890169
// MI455X (gfx1250) — matrix operand order repaired and checked by reading, not compiled, not run on hardware
//
#include <hip/hip_runtime.h>

typedef __attribute__((ext_vector_type(16))) __bf16 bf16x16;
typedef __attribute__((ext_vector_type(8)))  __bf16 bf16x8;
typedef __attribute__((ext_vector_type(8)))  float  f32x8;

#define WMMA_BF16(a, b, c) \
  __builtin_amdgcn_wmma_f32_16x16x32_bf16(false, (a), false, (b), (short)0, (c), false, false)

// ---------------------------------------------------------------------------
// Async global -> LDS copy, 16 bytes per lane (GLOBAL_LOAD_ASYNC_TO_LDS_B128,
// tracked by ASYNCcnt).  lds_byte_off is the per-lane LDS byte address.
// ---------------------------------------------------------------------------
__device__ __forceinline__ void async_cp16(unsigned lds_byte_off, const void* gsrc) {
  asm volatile("global_load_async_to_lds_b128 %0, %1, off"
               :: "v"(lds_byte_off), "v"((unsigned long long)(uintptr_t)gsrc)
               : "memory");
}
__device__ __forceinline__ void async_wait0() {
  asm volatile("s_wait_asynccnt 0" ::: "memory");
}

// ---------------------------------------------------------------------------
// A-operand fragment: 16x32 bf16 from LDS (row stride ldk).
// Lanes 0-15: M=lane, K{0..7,16..23}; lanes 16-31: M=lane-16, K{8..15,24..31}.
// ---------------------------------------------------------------------------
__device__ __forceinline__ bf16x16 frag_a(const __bf16* base, int ldk, int row0, int k0) {
  const int lane = threadIdx.x & 31;
  const int r    = row0 + (lane & 15);
  const int koff = (lane >> 4) << 3;   // 0 or 8
  const __bf16* p = base + r * ldk + k0 + koff;
  bf16x8 lo = *(const bf16x8*)p;
  bf16x8 hi = *(const bf16x8*)(p + 16);
  bf16x16 f;
#pragma unroll
  for (int i = 0; i < 8; ++i) { f[i] = lo[i]; f[i + 8] = hi[i]; }
  return f;
}

// ---------------------------------------------------------------------------
// B-operand fragment: 32x16 bf16 (K x N); source rows are the output-column
// entity (n) with contraction contiguous.  Lanes 0-15: col=lane, K{0..7,16..23};
// lanes 16-31: col=lane-16, K{8..15,24..31}.
// ---------------------------------------------------------------------------
__device__ __forceinline__ bf16x16 frag_b(const __bf16* base, int ldk, int col0, int k0) {
  const int lane = threadIdx.x & 31;
  const int n    = col0 + (lane & 15);
  const int koff = (lane >> 4) << 3;   // 0 or 8
  const __bf16* p = base + n * ldk + k0 + koff;
  bf16x8 lo = *(const bf16x8*)p;
  bf16x8 hi = *(const bf16x8*)(p + 16);
  bf16x16 f;
#pragma unroll
  for (int i = 0; i < 8; ++i) { f[i] = lo[i]; f[i + 8] = hi[i]; }
  return f;
}

// ---------------------------------------------------------------------------
// Kernel 1: fused QKV projection.  y = X @ W^T + b  (scaled for Q).
// grid = (N/128, M/64, 3), block = 256 (8 waves).  K-step 64, double-buffered
// LDS so stage i+1 global->LDS overlaps stage i WMMAs.
// Q,K -> bf16 [B][H][S][D]; V -> bf16 [B][H][D][S] (transposed for PV matmul).
// ---------------------------------------------------------------------------
__global__ __launch_bounds__(256) void qkv_proj_kernel(
    const float* __restrict__ X,
    const float* __restrict__ qw, const float* __restrict__ qb,
    const float* __restrict__ kw, const float* __restrict__ kb,
    const float* __restrict__ vw, const float* __restrict__ vb,
    __bf16* __restrict__ Qo, __bf16* __restrict__ Ko, __bf16* __restrict__ Vt) {
  constexpr int E = 1024, S = 2048, H = 16, D = 64;
  __shared__ __align__(16) __bf16 Xs[2][64 * 64];
  __shared__ __align__(16) __bf16 Ws[2][128 * 64];

  const int z = blockIdx.z;
  const float* W    = (z == 0) ? qw : (z == 1) ? kw : vw;
  const float* bias = (z == 0) ? qb : (z == 1) ? kb : vb;
  __bf16*      out  = (z == 0) ? Qo : (z == 1) ? Ko : Vt;
  const float scale = (z == 0) ? 0.125f : 1.0f;   // HEAD_DIM^-0.5 = 1/8

  const int tid = threadIdx.x;
  const int lane = tid & 31, wid = tid >> 5;
  const int mg = wid & 3;          // 4 row-groups of 16
  const int ng = wid >> 2;         // 2 col-groups of 64
  const int m0 = blockIdx.y * 64;
  const int n0 = blockIdx.x * 128;

  // Stage one 64-wide K slice of X (64 rows) and W (128 rows) into LDS buf,
  // converting fp32 -> bf16.  Fully unrolled: 2 X-chunks + 4 W-chunks/thread.
  auto stage = [&](int kk, int buf) {
#pragma unroll
    for (int i = 0; i < 2; ++i) {               // X tile 64x64
      const int c = tid + i * 256;
      const int row = c >> 3, cc = (c & 7) * 8;
      const float* src = X + (size_t)(m0 + row) * E + kk + cc;
      float4 f0 = *(const float4*)src;
      float4 f1 = *(const float4*)(src + 4);
      __bf16* dst = &Xs[buf][row * 64 + cc];
      dst[0] = (__bf16)f0.x; dst[1] = (__bf16)f0.y; dst[2] = (__bf16)f0.z; dst[3] = (__bf16)f0.w;
      dst[4] = (__bf16)f1.x; dst[5] = (__bf16)f1.y; dst[6] = (__bf16)f1.z; dst[7] = (__bf16)f1.w;
    }
#pragma unroll
    for (int i = 0; i < 4; ++i) {               // W tile 128x64
      const int c = tid + i * 256;
      const int row = c >> 3, cc = (c & 7) * 8;
      const float* src = W + (size_t)(n0 + row) * E + kk + cc;
      float4 f0 = *(const float4*)src;
      float4 f1 = *(const float4*)(src + 4);
      __bf16* dst = &Ws[buf][row * 64 + cc];
      dst[0] = (__bf16)f0.x; dst[1] = (__bf16)f0.y; dst[2] = (__bf16)f0.z; dst[3] = (__bf16)f0.w;
      dst[4] = (__bf16)f1.x; dst[5] = (__bf16)f1.y; dst[6] = (__bf16)f1.z; dst[7] = (__bf16)f1.w;
    }
  };

  f32x8 acc[4] = {};
  stage(0, 0);
  int buf = 0;

  for (int kk = 0; kk < E; kk += 64) {
    __syncthreads();
    if (kk + 64 < E) {
      stage(kk + 64, buf ^ 1);
      if (kk + 128 < E) {   // prime GL2 for the stage after next
        __builtin_prefetch(X + (size_t)(m0 + (tid >> 3)) * E + kk + 128, 0, 0);
        __builtin_prefetch(W + (size_t)(n0 + (tid >> 3)) * E + kk + 128, 0, 0);
      }
    }
#pragma unroll
    for (int kd = 0; kd < 64; kd += 32) {
      bf16x16 a = frag_a(Xs[buf], 64, mg * 16, kd);
#pragma unroll
      for (int t = 0; t < 4; ++t) {
        bf16x16 b = frag_b(Ws[buf], 64, ng * 64 + t * 16, kd);
        acc[t] = WMMA_BF16(a, b, acc[t]);
      }
    }
    buf ^= 1;
  }

  const int hi8 = (lane >> 4) << 3;
#pragma unroll
  for (int t = 0; t < 4; ++t) {
    const int n = n0 + ng * 64 + t * 16 + (lane & 15);
    const float bv = bias[n];
    const int h = n >> 6, d = n & 63;
#pragma unroll
    for (int r = 0; r < 8; ++r) {
      const int m = m0 + mg * 16 + r + hi8;
      const int b = m >> 11, s = m & 2047;
      const float y = (acc[t][r] + bv) * scale;
      size_t idx;
      if (z == 2) idx = ((size_t)(b * H + h) * D + d) * S + s;   // V transposed
      else        idx = ((size_t)(b * H + h) * S + s) * D + d;
      out[idx] = (__bf16)y;
    }
  }
}

// ---------------------------------------------------------------------------
// Kernel 2: flash-attention per (b*h, 64-query tile) + sum over queries.
// grid = (S/64, B*H), block = 256 (8 waves = 4 row-groups x 2 col-groups).
// All shared memory lives in one struct at LDS offset 0 so async-to-LDS
// destinations can be computed with __builtin_offsetof.
// ---------------------------------------------------------------------------
struct AttnSmem {
  __bf16 Qs[64 * 64];
  __bf16 Ks[64 * 64];
  __bf16 Vs[64 * 64];   // [d][key]
  __bf16 Ps[64 * 64];   // probs bf16
  float  Ss[64 * 64];   // scores fp32
  float  rowm[64], rowl[64], rowc[64], blocksum[64];
};

__global__ __launch_bounds__(256) void attn_sum_kernel(
    const __bf16* __restrict__ Q, const __bf16* __restrict__ K,
    const __bf16* __restrict__ Vt, float* __restrict__ out) {
  constexpr int S = 2048, D = 64;
  constexpr unsigned QS_OFF = __builtin_offsetof(AttnSmem, Qs);
  constexpr unsigned KS_OFF = __builtin_offsetof(AttnSmem, Ks);
  constexpr unsigned VS_OFF = __builtin_offsetof(AttnSmem, Vs);
  __shared__ __align__(16) AttnSmem sm;

  const int tid = threadIdx.x, lane = tid & 31, wid = tid >> 5;
  const int mg = wid >> 1, cg = wid & 1;
  const int q0 = blockIdx.x * 64;
  const int bh = blockIdx.y;
  const int hi8 = (lane >> 4) << 3;

  {   // Q tile: 4096 contiguous bf16, async copied 16B/lane/chunk
    const __bf16* base = Q + ((size_t)bh * S + q0) * D;
    async_cp16(QS_OFF + tid * 16,         base + tid * 8);
    async_cp16(QS_OFF + (tid + 256) * 16, base + (tid + 256) * 8);
  }
  if (tid < 64) { sm.rowm[tid] = -3.0e38f; sm.rowl[tid] = 0.0f; sm.blocksum[tid] = 0.0f; }

  f32x8 o0 = {}, o1 = {};

  for (int t0 = 0; t0 < S; t0 += 64) {
    {   // K tile: contiguous
      const __bf16* base = K + ((size_t)bh * S + t0) * D;
      async_cp16(KS_OFF + tid * 16,         base + tid * 8);
      async_cp16(KS_OFF + (tid + 256) * 16, base + (tid + 256) * 8);
    }
    {   // V^T tile: 64 rows (d), 64 keys contiguous per row
      const int row = tid >> 2, cc = (tid & 3) * 16;
      const __bf16* src = Vt + ((size_t)bh * D + row) * S + t0 + cc;
      const unsigned dsto = VS_OFF + (unsigned)(row * 64 + cc) * 2;
      async_cp16(dsto,      src);
      async_cp16(dsto + 16, src + 8);
    }
    async_wait0();
    __syncthreads();

    // scores tile 64x64 via WMMA (Q already carries the 1/sqrt(D) scale)
    f32x8 c0 = {}, c1 = {};
#pragma unroll
    for (int kd = 0; kd < 64; kd += 32) {
      bf16x16 a  = frag_a(sm.Qs, 64, mg * 16, kd);
      bf16x16 b0 = frag_b(sm.Ks, 64, cg * 32, kd);
      bf16x16 b1 = frag_b(sm.Ks, 64, cg * 32 + 16, kd);
      c0 = WMMA_BF16(a, b0, c0);
      c1 = WMMA_BF16(a, b1, c1);
    }
#pragma unroll
    for (int r = 0; r < 8; ++r) {
      const int row = mg * 16 + r + hi8;
      sm.Ss[row * 64 + cg * 32 + (lane & 15)]      = c0[r];
      sm.Ss[row * 64 + cg * 32 + 16 + (lane & 15)] = c1[r];
    }
    __syncthreads();

    if (tid < 64) {   // online softmax, one query row per thread
      const float mold = sm.rowm[tid];
      float tmax = -3.0e38f;
      const float* srow = sm.Ss + tid * 64;
      for (int j = 0; j < 64; ++j) tmax = fmaxf(tmax, srow[j]);
      const float mnew = fmaxf(mold, tmax);
      const float corr = __expf(mold - mnew);
      float sum = 0.0f;
      __bf16* prow = sm.Ps + tid * 64;
      for (int j = 0; j < 64; ++j) {
        const float p = __expf(srow[j] - mnew);
        sum += p;
        prow[j] = (__bf16)p;
      }
      sm.rowl[tid] = sm.rowl[tid] * corr + sum;
      sm.rowc[tid] = corr;
      sm.rowm[tid] = mnew;
    }
    __syncthreads();

    // rescale running O, then O += P @ V
#pragma unroll
    for (int r = 0; r < 8; ++r) {
      const float cf = sm.rowc[mg * 16 + r + hi8];
      o0[r] *= cf;
      o1[r] *= cf;
    }
#pragma unroll
    for (int kd = 0; kd < 64; kd += 32) {
      bf16x16 a  = frag_a(sm.Ps, 64, mg * 16, kd);
      bf16x16 b0 = frag_b(sm.Vs, 64, cg * 32, kd);
      bf16x16 b1 = frag_b(sm.Vs, 64, cg * 32 + 16, kd);
      o0 = WMMA_BF16(a, b0, o0);
      o1 = WMMA_BF16(a, b1, o1);
    }
    __syncthreads();   // protects Ks/Vs/Ps before next tile's async writes
  }

  // finalize: O /= l, sum over the 64 query rows, accumulate into out[bh][d]
  float p0 = 0.0f, p1 = 0.0f;
#pragma unroll
  for (int r = 0; r < 8; ++r) {
    const float inv = 1.0f / sm.rowl[mg * 16 + r + hi8];
    p0 += o0[r] * inv;
    p1 += o1[r] * inv;
  }
  p0 += __shfl_xor(p0, 16, 32);   // fold lanes16-31 rows into lanes0-15
  p1 += __shfl_xor(p1, 16, 32);
  if (lane < 16) {
    atomicAdd(&sm.blocksum[cg * 32 + lane],      p0);
    atomicAdd(&sm.blocksum[cg * 32 + 16 + lane], p1);
  }
  __syncthreads();
  if (tid < 64) atomicAdd(&out[bh * 64 + tid], sm.blocksum[tid]);
}

// ---------------------------------------------------------------------------
extern "C" void kernel_launch(void* const* d_in, const int* in_sizes, int n_in,
                              void* d_out, int out_size, void* d_ws, size_t ws_size,
                              hipStream_t stream) {
  (void)in_sizes; (void)n_in; (void)ws_size;
  const float* X  = (const float*)d_in[0];
  const float* qw = (const float*)d_in[1];
  const float* qb = (const float*)d_in[2];
  const float* kw = (const float*)d_in[3];
  const float* kb = (const float*)d_in[4];
  const float* vw = (const float*)d_in[5];
  const float* vb = (const float*)d_in[6];

  const size_t QKV_ELEMS = (size_t)4 * 16 * 2048 * 64;   // 8,388,608 bf16 each
  __bf16* Q  = (__bf16*)d_ws;
  __bf16* K  = Q + QKV_ELEMS;
  __bf16* Vt = K + QKV_ELEMS;

  hipMemsetAsync(d_out, 0, (size_t)out_size * sizeof(float), stream);

  qkv_proj_kernel<<<dim3(8, 128, 3), 256, 0, stream>>>(X, qw, qb, kw, kb, vw, vb, Q, K, Vt);
  attn_sum_kernel<<<dim3(32, 64), 256, 0, stream>>>(Q, K, Vt, (float*)d_out);
}
